// TransformerLayer4Dtransformer_15152644620348
// MI455X (gfx1250) — matrix operand order repaired and checked by reading, not compiled, not run on hardware
//
#include <hip/hip_runtime.h>
#include <hip/hip_bf16.h>

// Problem constants (fixed by reference).
#define B_   8
#define S_   1024
#define D_   512
#define H_   8
#define DK_  64
#define M_   (B_ * S_)          // 8192 rows in the projection GEMMs
#define NEG_BIG (-3.0e38f)

typedef __attribute__((ext_vector_type(16))) __bf16 v16bf;
typedef __attribute__((ext_vector_type(8)))  __bf16 v8bf;
typedef __attribute__((ext_vector_type(8)))  float  v8f;

// ---------------------------------------------------------------------------
// WMMA operand loaders (16x16x32 bf16; this code's contiguous k order, same for A and B):
//  A (16xK): lane L<16 holds row M=L with K elems 0..15 contiguous;
//            lane L+16 holds row M=L with K elems 16..31 contiguous.
//  B (Kx16): lane n<16 holds column n, K=0..15 contiguous;
//            lane n+16 holds column n, K=16..31 contiguous.
// Every operand here is a row of a row-major matrix, so loads are contiguous.
// ---------------------------------------------------------------------------
__device__ __forceinline__ v16bf load_a_tile(const __bf16* rowk, int lane) {
  const int off = ((lane >> 4) & 1) * 16;
  v8bf lo = *(const v8bf*)(rowk + off);
  v8bf hi = *(const v8bf*)(rowk + off + 8);
  return __builtin_shufflevector(lo, hi, 0, 1, 2, 3, 4, 5, 6, 7,
                                 8, 9, 10, 11, 12, 13, 14, 15);
}

__device__ __forceinline__ v16bf load_b_tile(const __bf16* colk, int lane) {
  const int off = ((lane >> 4) & 1) * 16;
  return *(const v16bf*)(colk + off);
}

// ---------------------------------------------------------------------------
// K1: fp32 -> bf16 conversion
// ---------------------------------------------------------------------------
__global__ void f2bf_kernel(const float* __restrict__ in,
                            __bf16* __restrict__ out, int n) {
  int i = blockIdx.x * blockDim.x + threadIdx.x;
  if (i < n) out[i] = (__bf16)in[i];
}

// ---------------------------------------------------------------------------
// K2: O[m,n] = sum_k X[m,k] * W[n,k] + bias[n]   (X:[M,K], W:[N,K] row-major)
// One wave computes a 16x64 strip (A reused across 4 WMMA tiles).
// mode 0: bf16 out, [B,H,S,dk] layout (q/k).  mode 1: bf16 out, [B,H,dk,S]
// (v transposed).  mode 2: fp32 out, plain row-major (output projection).
// ---------------------------------------------------------------------------
__global__ __launch_bounds__(256) void gemm_xwT_kernel(
    const __bf16* __restrict__ X, const __bf16* __restrict__ W,
    const float* __restrict__ bias, __bf16* __restrict__ outb,
    float* __restrict__ outf, int M, int N, int K, int mode) {
  const int lane = threadIdx.x & 31;
  const int strip = blockIdx.x * (blockDim.x >> 5) + (threadIdx.x >> 5);
  const int nStrips = N >> 6;
  const int m0 = (strip / nStrips) << 4;
  const int n0 = (strip % nStrips) << 6;
  if (m0 >= M) return;

  const int r = lane & 15;
  const __bf16* aRow = X + (size_t)(m0 + r) * K;

  v8f acc[4];
  for (int t = 0; t < 4; ++t) {
    const float bv = bias[n0 + t * 16 + r];
    for (int e = 0; e < 8; ++e) acc[t][e] = bv;
  }

  for (int kk = 0; kk < K; kk += 32) {
    if (kk + 128 < K) __builtin_prefetch(aRow + kk + 128, 0, 0);
    v16bf a = load_a_tile(aRow + kk, lane);
#pragma unroll
    for (int t = 0; t < 4; ++t) {
      const __bf16* wRow = W + (size_t)(n0 + t * 16 + r) * K + kk;
      v16bf b = load_b_tile(wRow, lane);
      acc[t] = __builtin_amdgcn_wmma_f32_16x16x32_bf16(
          false, a, false, b, (short)0, acc[t], false, false);
    }
  }

  // C/D layout: VGPR rr -> M = m0 + rr (+8 for upper half-wave), N = lane&15.
  const int mhi = (lane >> 4) & 1;
  for (int t = 0; t < 4; ++t) {
    for (int rr = 0; rr < 8; ++rr) {
      const int mrow = m0 + rr + mhi * 8;
      const int ncol = n0 + t * 16 + r;
      if (mode == 2) {
        outf[(size_t)mrow * N + ncol] = acc[t][rr];
      } else {
        const int b = mrow / S_, s = mrow % S_;
        const int h = ncol / DK_, d = ncol % DK_;
        const size_t idx =
            (mode == 0) ? ((((size_t)b * H_ + h) * S_ + s) * DK_ + d)
                        : ((((size_t)b * H_ + h) * DK_ + d) * S_ + s);
        outb[idx] = (__bf16)acc[t][rr];
      }
    }
  }
}

// ---------------------------------------------------------------------------
// K3: scores = (Q K^T) / sqrt(dk), causal tiles only, written into d_out's
// attn region (overwritten in place by the softmax kernel).
// One wave: 16 q-rows x 128 k-cols (A loaded once, up to 16 WMMAs).
// ---------------------------------------------------------------------------
__global__ __launch_bounds__(256) void qk_scores_kernel(
    const __bf16* __restrict__ Qb, const __bf16* __restrict__ Kb,
    float* __restrict__ scores) {
  const int lane = threadIdx.x & 31;
  const int g = blockIdx.x * (blockDim.x >> 5) + (threadIdx.x >> 5);
  const int bh = g >> 9;                 // 512 strips per (b,h)
  if (bh >= B_ * H_) return;
  const int rem = g & 511;
  const int q0 = (rem >> 3) << 4;        // 64 q-tiles
  const int k0s = (rem & 7) << 7;        // 8 k-strips of 128
  if (k0s > q0 + 15) return;             // strip fully above the diagonal

  const int r = lane & 15;
  const __bf16* Qh = Qb + (size_t)bh * S_ * DK_;
  const __bf16* Kh = Kb + (size_t)bh * S_ * DK_;
  const __bf16* qrow = Qh + (size_t)(q0 + r) * DK_;
  v16bf a0 = load_a_tile(qrow, lane);
  v16bf a1 = load_a_tile(qrow + 32, lane);

  const int mhi = (lane >> 4) & 1;
  for (int t = 0; t < 8; ++t) {
    const int k0 = k0s + t * 16;
    if (k0 > q0 + 15) break;             // remaining tiles are masked
    const __bf16* krow = Kh + (size_t)(k0 + r) * DK_;
    v8f acc;
    for (int e = 0; e < 8; ++e) acc[e] = 0.0f;
    acc = __builtin_amdgcn_wmma_f32_16x16x32_bf16(
        false, a0, false, load_b_tile(krow, lane), (short)0, acc, false, false);
    acc = __builtin_amdgcn_wmma_f32_16x16x32_bf16(
        false, a1, false, load_b_tile(krow + 32, lane), (short)0, acc, false,
        false);
    for (int rr = 0; rr < 8; ++rr) {
      const int qq = q0 + rr + mhi * 8;
      scores[((size_t)bh * S_ + qq) * S_ + k0 + r] = acc[rr] * 0.125f;
    }
  }
}

// ---------------------------------------------------------------------------
// K4: per-row softmax -> cumsum -> distance-decay -> softmax, in place.
// One 256-thread block per (b,h,q) row; 4 contiguous columns per thread.
// Masked (k > q) entries end as exact 0 (consumed by attn@V and the output).
// ---------------------------------------------------------------------------
__global__ __launch_bounds__(256) void softmax_decay_kernel(
    float* __restrict__ attn, const float* __restrict__ gammas) {
  const int row = blockIdx.x;            // B*H*S rows
  const int q = row & (S_ - 1);
  const int h = (row >> 10) & (H_ - 1);
  float* rowp = attn + (size_t)row * S_;
  const int tid = threadIdx.x;
  const int c0 = tid * 4;

  __shared__ float sd[256];

  float4 sv = *(const float4*)(rowp + c0);
  float sc[4] = {sv.x, sv.y, sv.z, sv.w};
  bool msk[4];
  for (int i = 0; i < 4; ++i) msk[i] = (c0 + i) <= q;

  // ---- softmax #1 ----
  float lmax = NEG_BIG;
  for (int i = 0; i < 4; ++i)
    if (msk[i]) lmax = fmaxf(lmax, sc[i]);
  sd[tid] = lmax;
  __syncthreads();
  for (int o = 128; o > 0; o >>= 1) {
    if (tid < o) sd[tid] = fmaxf(sd[tid], sd[tid + o]);
    __syncthreads();
  }
  float rmax = sd[0];
  __syncthreads();

  float e[4], lsum = 0.0f;
  for (int i = 0; i < 4; ++i) {
    e[i] = msk[i] ? __expf(sc[i] - rmax) : 0.0f;
    lsum += e[i];
  }
  sd[tid] = lsum;
  __syncthreads();
  for (int o = 128; o > 0; o >>= 1) {
    if (tid < o) sd[tid] += sd[tid + o];
    __syncthreads();
  }
  float inv = 1.0f / sd[0];
  __syncthreads();

  // ---- inclusive cumsum of probs (block-wide Hillis-Steele scan) ----
  float p[4], l[4], run = 0.0f;
  for (int i = 0; i < 4; ++i) {
    p[i] = e[i] * inv;
    run += p[i];
    l[i] = run;
  }
  const float tsum = run;
  sd[tid] = tsum;
  __syncthreads();
  for (int o = 1; o < 256; o <<= 1) {
    float v = (tid >= o) ? sd[tid - o] : 0.0f;
    __syncthreads();
    sd[tid] += v;
    __syncthreads();
  }
  const float incl = sd[tid];
  const float total = sd[255];
  const float excl = incl - tsum;
  __syncthreads();

  // ---- distance-decay rescale of the raw scores ----
  const float gam = -fabsf(gammas[h]);
  float s2[4];
  for (int i = 0; i < 4; ++i) {
    const float cum = excl + l[i];
    const float pos = fabsf((float)(q - (c0 + i)));
    float dd = (total - cum) * pos;
    dd = dd > 0.0f ? dd : 0.0f;
    float eff = __expf(sqrtf(dd) * gam);
    eff = fminf(fmaxf(eff, 1e-5f), 1e5f);
    s2[i] = sc[i] * eff;
  }

  // ---- softmax #2 ----
  lmax = NEG_BIG;
  for (int i = 0; i < 4; ++i)
    if (msk[i]) lmax = fmaxf(lmax, s2[i]);
  sd[tid] = lmax;
  __syncthreads();
  for (int o = 128; o > 0; o >>= 1) {
    if (tid < o) sd[tid] = fmaxf(sd[tid], sd[tid + o]);
    __syncthreads();
  }
  rmax = sd[0];
  __syncthreads();
  lsum = 0.0f;
  for (int i = 0; i < 4; ++i) {
    e[i] = msk[i] ? __expf(s2[i] - rmax) : 0.0f;
    lsum += e[i];
  }
  sd[tid] = lsum;
  __syncthreads();
  for (int o = 128; o > 0; o >>= 1) {
    if (tid < o) sd[tid] += sd[tid + o];
    __syncthreads();
  }
  inv = 1.0f / sd[0];

  float4 outv;
  outv.x = msk[0] ? e[0] * inv : 0.0f;
  outv.y = msk[1] ? e[1] * inv : 0.0f;
  outv.z = msk[2] ? e[2] * inv : 0.0f;
  outv.w = msk[3] ? e[3] * inv : 0.0f;
  *(float4*)(rowp + c0) = outv;
}

// ---------------------------------------------------------------------------
// K5: ctx = attn @ V.  A = attn rows (fp32 -> bf16 on the fly), B columns are
// rows of pre-transposed V^T [B,H,dk,S] (contiguous). Causal bound on k-loop.
// One wave: 16 q-rows x 64 cols (whole head width). Output in concat layout.
// ---------------------------------------------------------------------------
__global__ __launch_bounds__(256) void av_kernel(
    const float* __restrict__ attn, const __bf16* __restrict__ vT,
    __bf16* __restrict__ ctx) {
  const int lane = threadIdx.x & 31;
  const int g = blockIdx.x * (blockDim.x >> 5) + (threadIdx.x >> 5);
  const int bh = g >> 6;                 // 64 q-tiles per (b,h)
  if (bh >= B_ * H_) return;
  const int q0 = (g & 63) << 4;
  const int b = bh >> 3, h = bh & 7;
  const int r = lane & 15;
  const int off = ((lane >> 4) & 1) * 16;
  const int boff = ((lane >> 4) & 1) * 16;

  const float* arow = attn + ((size_t)bh * S_ + (q0 + r)) * S_;
  const __bf16* vh = vT + (size_t)bh * DK_ * S_;

  v8f acc[4];
  for (int t = 0; t < 4; ++t)
    for (int e2 = 0; e2 < 8; ++e2) acc[t][e2] = 0.0f;

  const int kend = q0 + 16;              // attn is exactly zero beyond q
  for (int kk = 0; kk < kend; kk += 32) {
    v8f lo = *(const v8f*)(arow + kk + off);
    v8f hi = *(const v8f*)(arow + kk + off + 8);
    v16bf a;
    for (int e2 = 0; e2 < 8; ++e2) {
      a[e2] = (__bf16)lo[e2];
      a[8 + e2] = (__bf16)hi[e2];
    }
#pragma unroll
    for (int t = 0; t < 4; ++t) {
      const __bf16* brow = vh + (size_t)(t * 16 + r) * S_ + kk + boff;
      v16bf bm = *(const v16bf*)brow;
      acc[t] = __builtin_amdgcn_wmma_f32_16x16x32_bf16(
          false, a, false, bm, (short)0, acc[t], false, false);
    }
  }

  const int mhi = (lane >> 4) & 1;
  for (int t = 0; t < 4; ++t) {
    for (int rr = 0; rr < 8; ++rr) {
      const int s = q0 + rr + mhi * 8;
      ctx[((size_t)b * S_ + s) * D_ + h * DK_ + t * 16 + r] =
          (__bf16)acc[t][rr];
    }
  }
}

// ---------------------------------------------------------------------------
// K6: y = LayerNorm(query + attn_out) * ln_w + ln_b.  One block per row.
// ---------------------------------------------------------------------------
__global__ __launch_bounds__(256) void resid_ln_kernel(
    const float* __restrict__ query, const float* __restrict__ ao,
    const float* __restrict__ lnw, const float* __restrict__ lnb,
    float* __restrict__ y) {
  const int row = blockIdx.x;            // B*S rows
  const int tid = threadIdx.x;
  __shared__ float sd[256];
  const float* qp = query + (size_t)row * D_;
  const float* ap = ao + (size_t)row * D_;
  const float x0 = qp[tid] + ap[tid];
  const float x1 = qp[tid + 256] + ap[tid + 256];

  sd[tid] = x0 + x1;
  __syncthreads();
  for (int o = 128; o > 0; o >>= 1) {
    if (tid < o) sd[tid] += sd[tid + o];
    __syncthreads();
  }
  const float mu = sd[0] * (1.0f / D_);
  __syncthreads();

  const float d0 = x0 - mu, d1 = x1 - mu;
  sd[tid] = d0 * d0 + d1 * d1;
  __syncthreads();
  for (int o = 128; o > 0; o >>= 1) {
    if (tid < o) sd[tid] += sd[tid + o];
    __syncthreads();
  }
  const float rinv = rsqrtf(sd[0] * (1.0f / D_) + 1e-5f);

  y[(size_t)row * D_ + tid] = d0 * rinv * lnw[tid] + lnb[tid];
  y[(size_t)row * D_ + tid + 256] = d1 * rinv * lnw[tid + 256] + lnb[tid + 256];
}

// ---------------------------------------------------------------------------
extern "C" void kernel_launch(void* const* d_in, const int* in_sizes, int n_in,
                              void* d_out, int out_size, void* d_ws,
                              size_t ws_size, hipStream_t stream) {
  (void)in_sizes; (void)n_in; (void)out_size; (void)ws_size;
  const float* query  = (const float*)d_in[0];
  const float* key    = (const float*)d_in[1];
  const float* values = (const float*)d_in[2];
  /* d_in[3] = lens: unused by the reference (mask is pure tril) */
  const float* Wq = (const float*)d_in[4];
  const float* bq = (const float*)d_in[5];
  const float* Wv = (const float*)d_in[6];
  const float* bv = (const float*)d_in[7];
  const float* Wo = (const float*)d_in[8];
  const float* bo = (const float*)d_in[9];
  const float* gammas = (const float*)d_in[10];
  const float* lnw = (const float*)d_in[11];
  const float* lnb = (const float*)d_in[12];

  float* out_y = (float*)d_out;                       // [B,S,D]
  float* out_attn = out_y + (size_t)B_ * S_ * D_;     // [B,H,S,S]

  // Workspace carve-up (~74 MB).
  char* ws = (char*)d_ws;
  size_t o = 0;
  auto take = [&](size_t bytes) -> char* {
    char* p = ws + o;
    o += (bytes + 255) & ~(size_t)255;
    return p;
  };
  __bf16* qb   = (__bf16*)take((size_t)B_ * H_ * S_ * DK_ * 2);  // [B,H,S,dk]
  __bf16* kb   = (__bf16*)take((size_t)B_ * H_ * S_ * DK_ * 2);
  __bf16* vtb  = (__bf16*)take((size_t)B_ * H_ * DK_ * S_ * 2);  // [B,H,dk,S]
  __bf16* xq   = (__bf16*)take((size_t)M_ * D_ * 2);
  __bf16* xk   = (__bf16*)take((size_t)M_ * D_ * 2);
  __bf16* xv   = (__bf16*)take((size_t)M_ * D_ * 2);
  __bf16* wqb  = (__bf16*)take((size_t)D_ * D_ * 2);
  __bf16* wvb  = (__bf16*)take((size_t)D_ * D_ * 2);
  __bf16* wob  = (__bf16*)take((size_t)D_ * D_ * 2);
  __bf16* ctxb = (__bf16*)take((size_t)M_ * D_ * 2);             // [B,S,D]
  float*  aout = (float*)take((size_t)M_ * D_ * 4);              // [B,S,D]

  const int nX = M_ * D_;    // 4,194,304
  const int nW = D_ * D_;    // 262,144
  f2bf_kernel<<<(nX + 255) / 256, 256, 0, stream>>>(query, xq, nX);
  f2bf_kernel<<<(nX + 255) / 256, 256, 0, stream>>>(key, xk, nX);
  f2bf_kernel<<<(nX + 255) / 256, 256, 0, stream>>>(values, xv, nX);
  f2bf_kernel<<<(nW + 255) / 256, 256, 0, stream>>>(Wq, wqb, nW);
  f2bf_kernel<<<(nW + 255) / 256, 256, 0, stream>>>(Wv, wvb, nW);
  f2bf_kernel<<<(nW + 255) / 256, 256, 0, stream>>>(Wo, wob, nW);

  // Projections: 4096 strips / 8 waves per block = 512 blocks.
  gemm_xwT_kernel<<<512, 256, 0, stream>>>(xq, wqb, bq, qb, nullptr,
                                           M_, D_, D_, 0);
  gemm_xwT_kernel<<<512, 256, 0, stream>>>(xk, wqb, bq, kb, nullptr,
                                           M_, D_, D_, 0);   // kq_same
  gemm_xwT_kernel<<<512, 256, 0, stream>>>(xv, wvb, bv, vtb, nullptr,
                                           M_, D_, D_, 1);   // V transposed

  // QK^T: 64*512 strips / 8 = 4096 blocks; scores land in d_out attn region.
  qk_scores_kernel<<<4096, 256, 0, stream>>>(qb, kb, out_attn);

  // In-place softmax / cumsum / decay / softmax: one block per row.
  softmax_decay_kernel<<<B_ * H_ * S_, 256, 0, stream>>>(out_attn, gammas);

  // attn @ V: 4096 waves / 8 = 512 blocks.
  av_kernel<<<512, 256, 0, stream>>>(out_attn, vtb, ctxb);

  // Output projection (fp32 out), then residual + LayerNorm.
  gemm_xwT_kernel<<<512, 256, 0, stream>>>(ctxb, wob, bo, nullptr, aout,
                                           M_, D_, D_, 2);
  resid_ln_kernel<<<M_, 256, 0, stream>>>(query, aout, lnw, lnb, out_y);
}
